// ExampleNet_37099927503448
// MI455X (gfx1250) — compile-verified, matrix operand order checked by reading, not run on hardware
//
#include <hip/hip_runtime.h>

// ---------------------------------------------------------------------------
// CDNA5 (gfx1250) implementation of the sparse-voxel 3D CNN.
// All 64->64 convolutions run as per-tap GEMMs on v_wmma_f32_16x16x32_f16:
//   M = 16 output voxels, K = input channels (2 chunks of 32), N = 16 out-ch
//   (4 groups to cover 64).
// Weights are repacked once per launch into per-lane B fragments, then staged
// into LDS per workgroup with the Tensor Data Mover (TENSOR_LOAD_TO_LDS,
// waited via s_wait_tensorcnt).  A fragments stream from global, software-
// pipelined one tap ahead of the WMMAs.
// ---------------------------------------------------------------------------

typedef __attribute__((ext_vector_type(16))) _Float16 v16h;
typedef __attribute__((ext_vector_type(8)))  _Float16 v8h;
typedef __attribute__((ext_vector_type(8)))  float    v8f;
typedef __attribute__((ext_vector_type(4)))  unsigned int u32x4;
typedef __attribute__((ext_vector_type(8)))  int      i32x8;
typedef __attribute__((ext_vector_type(4)))  int      i32x4;

#define NVOX31 59582   // 2*31*31*31
#define NVOX63 500094  // 2*63*63*63
#define NVOX61 453962  // 2*61*61*61

#define WP_HALFS (27 * 2 * 4 * 32 * 16)   // 110592 f16 per packed weight tensor
#define WP_BYTES (WP_HALFS * 2)           // 221184 B  (27648 x 8B)

// ---------------- workspace layout (bytes) ----------------
static constexpr size_t alup(size_t x) { return (x + 255) & ~size_t(255); }
static constexpr size_t OFF_X0 = 0;                              // 2*33^3*3 f32
static constexpr size_t SZ_X0  = alup(size_t(2*35937*3) * 4);
static constexpr size_t OFF_M0 = OFF_X0 + SZ_X0;                 // 2*33^3 f32
static constexpr size_t SZ_M0  = alup(size_t(2*35937) * 4);
static constexpr size_t OFF_X1 = OFF_M0 + SZ_M0;                 // 59584*64 f16
static constexpr size_t SZ_X   = alup(size_t(59584) * 64 * 2);
static constexpr size_t OFF_M1 = OFF_X1 + SZ_X;                  // 59584 u8
static constexpr size_t SZ_M1  = alup(size_t(59584));
static constexpr size_t OFF_X2 = OFF_M1 + SZ_M1;
static constexpr size_t OFF_X3 = OFF_X2 + SZ_X;
static constexpr size_t OFF_XT = OFF_X3 + SZ_X;                  // 500096*64 f16
static constexpr size_t SZ_XT  = alup(size_t(500096) * 64 * 2);
static constexpr size_t OFF_WP2 = OFF_XT + SZ_XT;                // 110592 f16 each
static constexpr size_t SZ_WP  = alup(size_t(WP_HALFS) * 2);
static constexpr size_t OFF_WP3 = OFF_WP2 + SZ_WP;
static constexpr size_t OFF_WPT = OFF_WP3 + SZ_WP;
static constexpr size_t OFF_WP5 = OFF_WPT + SZ_WP;
// total ~85 MiB of d_ws required

// ---------------- helpers ----------------
__device__ __forceinline__ void dec31(int vox, int& b, int& z, int& y, int& x) {
  b = vox / 29791; int r = vox - b * 29791;
  z = r / 961; r -= z * 961; y = r / 31; x = r - y * 31;
}
__device__ __forceinline__ void dec63(int vox, int& b, int& z, int& y, int& x) {
  b = vox / 250047; int r = vox - b * 250047;
  z = r / 3969; r -= z * 3969; y = r / 63; x = r - y * 63;
}
__device__ __forceinline__ void dec61(int vox, int& b, int& z, int& y, int& x) {
  b = vox / 226981; int r = vox - b * 226981;
  z = r / 3721; r -= z * 3721; y = r / 61; x = r - y * 61;
}

// A-fragment load for V_WMMA_F32_16X16X32_F16: per lane, elements 0..7 hold
// channels kbase+off+0..7 and elements 8..15 hold kbase+off+16..23, with
// off = 8*(lane>=16).  `base` points at the 64 f16 channels of this lane's
// voxel (M = lane&15).
__device__ __forceinline__ v16h load_frag_a(const _Float16* __restrict__ base,
                                            int kbase, int laneHi) {
  union { v16h v; v8h h[2]; } u;
  const _Float16* p = base + kbase + (laneHi ? 8 : 0);
  u.h[0] = *(const v8h*)(p);
  u.h[1] = *(const v8h*)(p + 16);
  return u.v;
}

// B-fragment read from LDS (16B-aligned accesses -> 2x ds_load_b128).
__device__ __forceinline__ v16h load_frag_b_lds(const _Float16* __restrict__ p) {
  union { v16h v; v8h h[2]; } u;
  u.h[0] = *(const v8h*)(p);
  u.h[1] = *(const v8h*)(p + 8);
  return u.v;
}

// Stage the 216KB packed-weight blob into LDS (offset 0 of dynamic LDS).
// Preferred path: one TDM descriptor issued by wave 0, 1-D tile of 27648
// 8-byte elements, completion via s_wait_tensorcnt; then block barrier.
__device__ __forceinline__ void stage_weights(const _Float16* __restrict__ wp,
                                              _Float16* __restrict__ lw) {
#if __has_builtin(__builtin_amdgcn_tensor_load_to_lds) && \
    __has_builtin(__builtin_amdgcn_s_wait_tensorcnt)
  if ((threadIdx.x >> 5) == 0) {
    unsigned long long ga = (unsigned long long)(const void*)wp;
    u32x4 g0;
    g0.x = 1u;                                   // count=1 (valid descriptor)
    g0.y = 0u;                                   // lds_addr = 0
    g0.z = (unsigned)ga;                         // global_addr[31:0]
    g0.w = (unsigned)((ga >> 32) & 0x1FFFFFFu)   // global_addr[56:32]
           | 0x80000000u;                        // type = 2 ("image")
    i32x8 g1;
    g1[0] = 0x30000;                 // data_size=3 (8B), workgroup_mask=0
    g1[1] = (int)(0x6C00u << 16);    // tensor_dim0 lo16 = 27648
    g1[2] = 0x00010000;              // tensor_dim0 hi16=0, tensor_dim1 lo16=1
    g1[3] = (int)(0x6C00u << 16);    // tensor_dim1 hi16=0, tile_dim0 = 27648
    g1[4] = 1;                       // tile_dim1 = 1, tile_dim2 = 0
    g1[5] = 0x6C00;                  // tensor_dim0_stride = 27648
    g1[6] = 0;
    g1[7] = 0;
    i32x4 gz = {0, 0, 0, 0};
#if __clang_major__ >= 23
    i32x8 gz8 = {0, 0, 0, 0, 0, 0, 0, 0};
    __builtin_amdgcn_tensor_load_to_lds(g0, g1, gz, gz, gz8, 0);
#else
    __builtin_amdgcn_tensor_load_to_lds(g0, g1, gz, gz, 0);
#endif
    __builtin_amdgcn_s_wait_tensorcnt(0);
  }
  __syncthreads();
  (void)lw;
#else
  for (int c = threadIdx.x; c < WP_BYTES / 16; c += blockDim.x)
    *(v8h*)(lw + c * 8) = *(const v8h*)(wp + c * 8);
  __syncthreads();
#endif
}

// ---------------- utility kernels ----------------
__global__ void k_zero(float* __restrict__ p, size_t n) {
  for (size_t i = (size_t)blockIdx.x * blockDim.x + threadIdx.x; i < n;
       i += (size_t)gridDim.x * blockDim.x)
    p[i] = 0.f;
}

__global__ void k_scatter(const float* __restrict__ feats,
                          const int* __restrict__ coors,
                          float* __restrict__ x0, float* __restrict__ m0, int n) {
  int i = blockIdx.x * blockDim.x + threadIdx.x;
  if (i >= n) return;
  int b = coors[i * 4 + 0], z = coors[i * 4 + 1];
  int y = coors[i * 4 + 2], x = coors[i * 4 + 3];
  int ii = ((b * 33 + z) * 33 + y) * 33 + x;
  x0[ii * 3 + 0] = feats[i * 3 + 0];
  x0[ii * 3 + 1] = feats[i * 3 + 1];
  x0[ii * 3 + 2] = feats[i * 3 + 2];
  m0[ii] = 1.f;
}

// Repack a (3,3,3,64,64) f32 weight tensor into per-lane WMMA B fragments:
// wp[(((t*2+q)*4+g)*32 + lane)*16 + e]  (f16, 32B contiguous per lane).
// flip=1 bakes in the spatial mirror needed for conv_transpose.
__global__ void k_repack(const float* __restrict__ w, _Float16* __restrict__ wp,
                         int flip) {
  int i = blockIdx.x * blockDim.x + threadIdx.x;
  if (i >= WP_HALFS) return;
  int e = i & 15, l = (i >> 4) & 31, g = (i >> 9) & 3, q = (i >> 11) & 1, t = i >> 12;
  int tw = flip ? (26 - t) : t;
  int n = g * 16 + (l & 15);
  int p = e >> 1;
  int k = 32 * q + ((p < 4) ? 2 * p : 16 + 2 * (p - 4)) + ((l & 16) ? 8 : 0) + (e & 1);
  wp[i] = (_Float16)w[(tw * 64 + k) * 64 + n];
}

// conv1: 3->64 VALID + mask + fused act.  Tiny FLOPs -> scalar VALU kernel.
__global__ void k_conv1(const float* __restrict__ x0, const float* __restrict__ m0,
                        const float* __restrict__ w1, const float* __restrict__ b1,
                        const float* __restrict__ s1, const float* __restrict__ h1,
                        _Float16* __restrict__ x1, unsigned char* __restrict__ m1) {
  int vox = blockIdx.x;
  int co = threadIdx.x;  // 64 threads
  int b, z, y, x; dec31(vox, b, z, y, x);
  float sum = 0.f, occ = 0.f;
  #pragma unroll
  for (int t = 0; t < 27; ++t) {
    int zz = z + t / 9, yy = y + (t / 3) % 3, xx = x + t % 3;
    int ii = ((b * 33 + zz) * 33 + yy) * 33 + xx;
    occ += m0[ii];
    #pragma unroll
    for (int ci = 0; ci < 3; ++ci)
      sum += x0[ii * 3 + ci] * w1[(t * 3 + ci) * 64 + co];
  }
  bool mm = occ > 0.f;
  float v = sum + b1[co];
  v = fmaxf(v * s1[co] + h1[co], 0.f);
  if (!mm) v = 0.f;
  x1[(size_t)vox * 64 + co] = (_Float16)v;
  if (co == 0) m1[vox] = mm ? 1 : 0;
}

// 64->64 SAME 3x3x3 conv over the 31^3 grid, fused masked activation.
// 8 waves per block, one 16-voxel tile per wave; weights in LDS.
__global__ __launch_bounds__(256) void k_conv_same_wmma(
    const _Float16* __restrict__ xin, const unsigned char* __restrict__ mask,
    const _Float16* __restrict__ wp, const float* __restrict__ bias,
    const float* __restrict__ sc, const float* __restrict__ sh,
    _Float16* __restrict__ xout) {
  extern __shared__ _Float16 lw[];
  stage_weights(wp, lw);

  const int lane = threadIdx.x & 31;
  const int tile = blockIdx.x * 8 + (threadIdx.x >> 5);
  const int base = tile * 16;
  const int laneHi = lane >> 4;
  const int m = lane & 15;
  const int vox = base + m;
  const bool valid = vox < NVOX31;
  int b, z, y, x; dec31(valid ? vox : 0, b, z, y, x);

  v8f acc[4];
  #pragma unroll
  for (int g = 0; g < 4; ++g) acc[g] = (v8f){};
  const v16h zf = {};

  auto loadA = [&](int t, v16h& a0, v16h& a1) {
    const int kz = t / 9 - 1, ky = (t / 3) % 3 - 1, kx = t % 3 - 1;
    const int zz = z + kz, yy = y + ky, xx = x + kx;
    const bool nv = valid && (unsigned)zz < 31u && (unsigned)yy < 31u && (unsigned)xx < 31u;
    const _Float16* ap = xin + (size_t)(nv ? ((b * 31 + zz) * 31 + yy) * 31 + xx : 0) * 64;
    a0 = nv ? load_frag_a(ap, 0, laneHi) : zf;
    a1 = nv ? load_frag_a(ap, 32, laneHi) : zf;
  };

  v16h a0, a1;
  loadA(0, a0, a1);
  #pragma unroll
  for (int t = 0; t < 27; ++t) {
    v16h n0 = zf, n1 = zf;
    if (t < 26) loadA(t + 1, n0, n1);  // overlap next-tap A with WMMAs
    #pragma unroll
    for (int q = 0; q < 2; ++q) {
      const v16h A = q ? a1 : a0;
      const _Float16* bp = lw + (size_t)(((t * 2 + q) * 4) * 32 + lane) * 16;
      #pragma unroll
      for (int g = 0; g < 4; ++g) {
        v16h B = load_frag_b_lds(bp + (size_t)g * 512);
        acc[g] = __builtin_amdgcn_wmma_f32_16x16x32_f16(
            false, A, false, B, (short)0, acc[g], false, false);
      }
    }
    a0 = n0; a1 = n1;
  }
  // epilogue: D layout => col N = lane&15, row M = r + 8*laneHi
  #pragma unroll
  for (int g = 0; g < 4; ++g) {
    const int n = g * 16 + m;
    const float bn = bias[n], sn = sc[n], hn = sh[n];
    #pragma unroll
    for (int r = 0; r < 8; ++r) {
      const int ov = base + r + 8 * laneHi;
      if (ov < NVOX31) {
        float v = acc[g][r] + bn;
        v = fmaxf(v * sn + hn, 0.f);
        if (!mask[ov]) v = 0.f;
        xout[(size_t)ov * 64 + n] = (_Float16)v;
      }
    }
  }
}

// 64->64 stride-2 conv-transpose (3x3x3, VALID) as an output-gather GEMM over
// the 63^3 grid.  lax.conv_transpose (no kernel flip) => out[z] += in[iz]*w[kz]
// with z = 2*iz + (2-kz); the flip is baked into the repacked weights.
__global__ __launch_bounds__(256) void k_convT_wmma(
    const _Float16* __restrict__ xin, const unsigned char* __restrict__ m1,
    const _Float16* __restrict__ wp, const float* __restrict__ bias,
    const float* __restrict__ sc, const float* __restrict__ sh,
    _Float16* __restrict__ xt) {
  extern __shared__ _Float16 lw[];
  stage_weights(wp, lw);

  const int lane = threadIdx.x & 31;
  const int tile = blockIdx.x * 8 + (threadIdx.x >> 5);
  const int base = tile * 16;
  const int laneHi = lane >> 4;
  const int m = lane & 15;
  const int vox = base + m;
  const bool valid = vox < NVOX63;
  int b, z, y, x; dec63(valid ? vox : 0, b, z, y, x);

  v8f acc[4];
  #pragma unroll
  for (int g = 0; g < 4; ++g) acc[g] = (v8f){};
  const v16h zf = {};
  int occ_own = 0;  // transposed mask for this lane's voxel (M = lane&15)

  auto loadA = [&](int t, v16h& a0, v16h& a1) {
    const int az = t / 9, ay = (t / 3) % 3, ax = t % 3;
    const int rz = z - az, ry = y - ay, rx = x - ax;
    bool nv = valid && rz >= 0 && ry >= 0 && rx >= 0 && !((rz | ry | rx) & 1);
    const int iz = rz >> 1, iy = ry >> 1, ix = rx >> 1;
    nv = nv && iz < 31 && iy < 31 && ix < 31;
    const int nidx = ((b * 31 + iz) * 31 + iy) * 31 + ix;
    if (nv && m1[nidx]) occ_own = 1;
    const _Float16* ap = xin + (size_t)(nv ? nidx : 0) * 64;
    a0 = nv ? load_frag_a(ap, 0, laneHi) : zf;
    a1 = nv ? load_frag_a(ap, 32, laneHi) : zf;
  };

  v16h a0, a1;
  loadA(0, a0, a1);
  #pragma unroll
  for (int t = 0; t < 27; ++t) {
    v16h n0 = zf, n1 = zf;
    if (t < 26) loadA(t + 1, n0, n1);
    #pragma unroll
    for (int q = 0; q < 2; ++q) {
      const v16h A = q ? a1 : a0;
      const _Float16* bp = lw + (size_t)(((t * 2 + q) * 4) * 32 + lane) * 16;
      #pragma unroll
      for (int g = 0; g < 4; ++g) {
        v16h B = load_frag_b_lds(bp + (size_t)g * 512);
        acc[g] = __builtin_amdgcn_wmma_f32_16x16x32_f16(
            false, A, false, B, (short)0, acc[g], false, false);
      }
    }
    a0 = n0; a1 = n1;
  }
  #pragma unroll
  for (int g = 0; g < 4; ++g) {
    const int n = g * 16 + m;
    const float bn = bias[n], sn = sc[n], hn = sh[n];
    #pragma unroll
    for (int r = 0; r < 8; ++r) {
      const int ov = base + r + 8 * laneHi;
      // fetch row-voxel's occupancy from the lane that owns it (ds_bpermute)
      const int occ_r = __shfl(occ_own, r + 8 * laneHi, 32);
      if (ov < NVOX63) {
        float v = acc[g][r] + bn;
        v = fmaxf(v * sn + hn, 0.f);
        if (!occ_r) v = 0.f;
        xt[(size_t)ov * 64 + n] = (_Float16)v;
      }
    }
  }
}

// 64->64 VALID 3x3x3 conv over 63^3 -> 61^3, fused relu(v*s+h), channel-first
// f32 output.  Dominant-FLOP stage (~1e11 FLOPs).
__global__ __launch_bounds__(256) void k_conv5_wmma(
    const _Float16* __restrict__ xin, const _Float16* __restrict__ wp,
    const float* __restrict__ bias, const float* __restrict__ sc,
    const float* __restrict__ sh, float* __restrict__ out) {
  extern __shared__ _Float16 lw[];
  stage_weights(wp, lw);

  const int lane = threadIdx.x & 31;
  const int tile = blockIdx.x * 8 + (threadIdx.x >> 5);
  const int base = tile * 16;
  const int laneHi = lane >> 4;
  const int m = lane & 15;
  const int vox = base + m;
  const bool valid = vox < NVOX61;
  int b, z, y, x; dec61(valid ? vox : 0, b, z, y, x);

  v8f acc[4];
  #pragma unroll
  for (int g = 0; g < 4; ++g) acc[g] = (v8f){};
  const v16h zf = {};

  auto loadA = [&](int t, v16h& a0, v16h& a1) {
    const int zz = z + t / 9, yy = y + (t / 3) % 3, xx = x + t % 3;  // in [0,63)
    const int nidx = ((b * 63 + zz) * 63 + yy) * 63 + xx;
    const _Float16* ap = xin + (size_t)(valid ? nidx : 0) * 64;
    a0 = valid ? load_frag_a(ap, 0, laneHi) : zf;
    a1 = valid ? load_frag_a(ap, 32, laneHi) : zf;
  };

  v16h a0, a1;
  loadA(0, a0, a1);
  #pragma unroll
  for (int t = 0; t < 27; ++t) {
    v16h n0 = zf, n1 = zf;
    if (t < 26) loadA(t + 1, n0, n1);
    #pragma unroll
    for (int q = 0; q < 2; ++q) {
      const v16h A = q ? a1 : a0;
      const _Float16* bp = lw + (size_t)(((t * 2 + q) * 4) * 32 + lane) * 16;
      #pragma unroll
      for (int g = 0; g < 4; ++g) {
        v16h B = load_frag_b_lds(bp + (size_t)g * 512);
        acc[g] = __builtin_amdgcn_wmma_f32_16x16x32_f16(
            false, A, false, B, (short)0, acc[g], false, false);
      }
    }
    a0 = n0; a1 = n1;
  }
  #pragma unroll
  for (int r = 0; r < 8; ++r) {
    const int ov = base + r + 8 * laneHi;
    if (ov < NVOX61) {
      int ob, oz, oy, ox; dec61(ov, ob, oz, oy, ox);
      #pragma unroll
      for (int g = 0; g < 4; ++g) {
        const int n = g * 16 + m;
        float v = acc[g][r] + bias[n];
        v = fmaxf(v * sc[n] + sh[n], 0.f);
        out[((size_t)(ob * 64 + n)) * 226981 + (size_t)oz * 3721 + oy * 61 + ox] = v;
      }
    }
  }
}

// ---------------------------------------------------------------------------
extern "C" void kernel_launch(void* const* d_in, const int* in_sizes, int n_in,
                              void* d_out, int out_size, void* d_ws, size_t ws_size,
                              hipStream_t stream) {
  (void)n_in; (void)out_size; (void)ws_size;  // requires ~85 MiB of d_ws
  const float* feats = (const float*)d_in[0];
  const int*   coors = (const int*)d_in[1];
  const float* w1 = (const float*)d_in[3];
  const float* b1 = (const float*)d_in[4];
  const float* w2 = (const float*)d_in[5];
  const float* b2 = (const float*)d_in[6];
  const float* w3 = (const float*)d_in[7];
  const float* b3 = (const float*)d_in[8];
  const float* wt = (const float*)d_in[9];
  const float* bt = (const float*)d_in[10];
  const float* w5 = (const float*)d_in[11];
  const float* b5 = (const float*)d_in[12];
  const float* s1 = (const float*)d_in[13];
  const float* h1 = (const float*)d_in[14];
  const float* s2 = (const float*)d_in[15];
  const float* h2 = (const float*)d_in[16];
  const float* s3 = (const float*)d_in[17];
  const float* h3 = (const float*)d_in[18];
  const float* s4 = (const float*)d_in[19];
  const float* h4 = (const float*)d_in[20];
  const float* s5 = (const float*)d_in[21];
  const float* h5 = (const float*)d_in[22];
  const int N = in_sizes[0] / 3;

  char* ws = (char*)d_ws;
  float*         x0  = (float*)(ws + OFF_X0);
  float*         m0  = (float*)(ws + OFF_M0);
  _Float16*      x1  = (_Float16*)(ws + OFF_X1);
  unsigned char* m1  = (unsigned char*)(ws + OFF_M1);
  _Float16*      x2  = (_Float16*)(ws + OFF_X2);
  _Float16*      x3  = (_Float16*)(ws + OFF_X3);
  _Float16*      xt  = (_Float16*)(ws + OFF_XT);
  _Float16*      wp2 = (_Float16*)(ws + OFF_WP2);
  _Float16*      wp3 = (_Float16*)(ws + OFF_WP3);
  _Float16*      wpt = (_Float16*)(ws + OFF_WPT);
  _Float16*      wp5 = (_Float16*)(ws + OFF_WP5);

  // dense input + mask must start from zero every call
  k_zero<<<512, 256, 0, stream>>>((float*)(ws + OFF_X0), (SZ_X0 + SZ_M0) / 4);
  k_scatter<<<(N + 255) / 256, 256, 0, stream>>>(feats, coors, x0, m0, N);

  const int RP = (WP_HALFS + 255) / 256;
  k_repack<<<RP, 256, 0, stream>>>(w2, wp2, 0);
  k_repack<<<RP, 256, 0, stream>>>(w3, wp3, 0);
  k_repack<<<RP, 256, 0, stream>>>(wt, wpt, 1);
  k_repack<<<RP, 256, 0, stream>>>(w5, wp5, 0);

  k_conv1<<<NVOX31, 64, 0, stream>>>(x0, m0, w1, b1, s1, h1, x1, m1);

  const int T31 = (NVOX31 + 15) / 16;  // 3724 tiles
  k_conv_same_wmma<<<(T31 + 7) / 8, 256, WP_BYTES, stream>>>(x1, m1, wp2, b2, s2, h2, x2);
  k_conv_same_wmma<<<(T31 + 7) / 8, 256, WP_BYTES, stream>>>(x2, m1, wp3, b3, s3, h3, x3);

  const int T63 = (NVOX63 + 15) / 16;  // 31256 tiles
  k_convT_wmma<<<(T63 + 7) / 8, 256, WP_BYTES, stream>>>(x3, m1, wpt, bt, s4, h4, xt);

  const int T61 = (NVOX61 + 15) / 16;  // 28373 tiles
  k_conv5_wmma<<<(T61 + 7) / 8, 256, WP_BYTES, stream>>>(xt, wp5, b5, s5, h5, (float*)d_out);
}
